// DIMKT_56049323212937
// MI455X (gfx1250) — matrix operand order repaired and checked by reading, not compiled, not run on hardware
//
#include <hip/hip_runtime.h>
#include <hip/hip_bf16.h>

#define BB 512
#define SS 500
#define DD 128

typedef __attribute__((ext_vector_type(16))) __bf16 v16bf;
typedef __attribute__((ext_vector_type(8)))  float  v8f;
typedef __attribute__((ext_vector_type(4)))  int    v4i;

#if __has_builtin(__builtin_amdgcn_global_load_async_to_lds_b128)
#define HAVE_ASYNC_LDS 1
#endif

__device__ __forceinline__ float sigmoidf_(float x) { return 1.0f / (1.0f + __expf(-x)); }

__device__ __forceinline__ v8f wmma_bf16(v16bf a, v16bf b, v8f c) {
  // 8 args: (neg_a, A, neg_b, B, c_mod, C, reuse_a, reuse_b)
  return __builtin_amdgcn_wmma_f32_16x16x32_bf16(false, a, false, b, (short)0, c, false, false);
}

#if HAVE_ASYNC_LDS
// Signature discovered from clang diagnostic: (int4 AS1*, int4 AS3*, imm, imm)
__device__ __forceinline__ void async_copy_b128(const void* gsrc, void* ldst) {
  __builtin_amdgcn_global_load_async_to_lds_b128(
      (__attribute__((address_space(1))) v4i*)(void*)gsrc,
      (__attribute__((address_space(3))) v4i*)(void*)ldst, 0, 0);
}
__device__ __forceinline__ void wait_async0() {
#if __has_builtin(__builtin_amdgcn_s_wait_asynccnt)
  __builtin_amdgcn_s_wait_asynccnt(0);
#else
  asm volatile("s_wait_asynccnt 0" ::: "memory");
#endif
}
#endif

// A-operand (16x32 bf16) swizzle: lane half by (kk>>3)&1, element index packs
// K pairs per VGPR per the ISA 16-bit A layout. tileIdx = mt*16 + kt.
__device__ __forceinline__ int a_index(int mt, int kt, int m15, int kk) {
  int half = (kk >> 3) & 1;
  int e = (kk & 7) + ((kk & 16) >> 1);  // +8 when kk>=16
  int lane = half * 16 + m15;
  return ((mt * 16 + kt) * 32 + lane) * 16 + e;
}

// ---------------------------------------------------------------------------
// Kernel 1: convert fp32 weight [K,128] (row-major, y = x@W) into bf16,
// pre-swizzled into the WMMA B-operand (32x16) tile layout:
//   tile (kt, nt): lane<16 -> N=lane,  K=kt*32+e (e<8), kt*32+16+(e-8) (e>=8)
//                  lane>=16 -> N=lane-16, K=kt*32+8+e (e<8), kt*32+24+(e-8) (e>=8)
// ---------------------------------------------------------------------------
__global__ __launch_bounds__(256) void swz_w(const float* __restrict__ W,
                                             __bf16* __restrict__ out, int Kdim) {
  int i = blockIdx.x * 256 + threadIdx.x;
  if (i >= Kdim * DD) return;
  int k = i >> 7, n = i & 127;
  int kt = k >> 5, kk = k & 31, ntile = n >> 4, ln = n & 15;
  int half = (kk >> 3) & 1, e = (kk & 7) + ((kk & 16) >> 1);
  int lane = half * 16 + ln;
  out[((kt * 8 + ntile) * 32 + lane) * 16 + e] = (__bf16)W[i];
}

// ---------------------------------------------------------------------------
// Kernel 2: x_all = concat(qe,ce,qde,cde) @ Wx + bx  ->  bf16 [B*S, D]
// 64-row tiles per WG, 8 waves: wave = N-column, 4 M-tiles each, K=512.
// ---------------------------------------------------------------------------
__global__ __launch_bounds__(256) void xgemm_kernel(
    const int* __restrict__ qs, const int* __restrict__ cs,
    const int* __restrict__ qds, const int* __restrict__ cds,
    const float* __restrict__ Eq, const float* __restrict__ Ec,
    const float* __restrict__ Eqd, const float* __restrict__ Ecd,
    const __bf16* __restrict__ wxs, const float* __restrict__ bx,
    __bf16* __restrict__ xw) {
  __shared__ __attribute__((aligned(64))) __bf16 act[64 * 512];
  int tid = threadIdx.x;
  int row0 = blockIdx.x * 64;

  // Gather the 4 embedding rows (concat K=512) into LDS, A-swizzled, bf16.
  {
    int m = tid >> 2;    // 0..63
    int seg = tid & 3;   // which table / 128-col block
    int g = row0 + m;
    const float* src;
    if (seg == 0)      src = Eq  + (size_t)qs[g]  * DD;
    else if (seg == 1) src = Ec  + (size_t)cs[g]  * DD;
    else if (seg == 2) src = Eqd + (size_t)qds[g] * DD;
    else               src = Ecd + (size_t)cds[g] * DD;
    int mt = m >> 4, m15 = m & 15;
    #pragma unroll 4
    for (int c = 0; c < DD; ++c) {
      int k = seg * DD + c;
      act[a_index(mt, k >> 5, m15, k & 31)] = (__bf16)src[c];
    }
  }
  __syncthreads();

  int wave = tid >> 5, lane = tid & 31;
  int nt = wave;
  v8f acc[4] = {};
  for (int kt = 0; kt < 16; ++kt) {
    v16bf b = *(const v16bf*)(wxs + ((kt * 8 + nt) * 32 + lane) * 16);
    #pragma unroll
    for (int mt = 0; mt < 4; ++mt) {
      v16bf a = *(const v16bf*)(act + ((mt * 16 + kt) * 32 + lane) * 16);
      acc[mt] = wmma_bf16(a, b, acc[mt]);
    }
  }
  int n = nt * 16 + (lane & 15);
  float bias = bx[n];
  int mhalf = (lane >= 16) ? 8 : 0;
  #pragma unroll
  for (int mt = 0; mt < 4; ++mt) {
    #pragma unroll
    for (int r = 0; r < 8; ++r) {
      int m = mt * 16 + r + mhalf;
      xw[(size_t)(row0 + m) * DD + n] = (__bf16)(acc[mt][r] + bias);
    }
  }
}

// ---------------------------------------------------------------------------
// Kernel 3: persistent recurrent scan. 32 WGs x 16 batch rows; 8 waves = 8
// N-columns of the [16,128] state tile. Weights streamed bf16 from L2 in the
// B-operand layout; activations restaged in LDS (A layout) between GEMMs;
// x_{t+1} tile async-prefetched into LDS and consumed by the y_t dot product.
// ---------------------------------------------------------------------------
__global__ __launch_bounds__(256) void scan_kernel(
    const __bf16* __restrict__ xw,
    const __bf16* __restrict__ wsdf1s, const __bf16* __restrict__ wsdf2s,
    const __bf16* __restrict__ wpka1s, const __bf16* __restrict__ wpka2s,
    const __bf16* __restrict__ wkis,
    const float* __restrict__ bsdf1, const float* __restrict__ bsdf2,
    const float* __restrict__ bpka1, const float* __restrict__ bpka2,
    const float* __restrict__ bki,
    const int* __restrict__ corr, const int* __restrict__ qds,
    const int* __restrict__ cds,
    const float* __restrict__ Ecorr, const float* __restrict__ Eqd,
    const float* __restrict__ Ecd,
    const float* __restrict__ h0, float* __restrict__ y) {
  __shared__ __attribute__((aligned(64))) __bf16 act[16 * 512];  // A staging
  __shared__ __attribute__((aligned(64))) float  hbuf[16 * DD];  // state h
  __shared__ __attribute__((aligned(64))) __bf16 xnb[16 * DD];   // x_{t+1} prefetch
  __shared__ float yred[16];

  int tid = threadIdx.x;
  int wave = tid >> 5, lane = tid & 31;
  int nt = wave;
  int n = nt * 16 + (lane & 15);
  int mhalf = (lane >= 16) ? 8 : 0;
  int r0 = blockIdx.x * 16;

  // init h from h0; zero the never-written last column of y
  {
    int m = tid >> 4, c = (tid & 15) * 8;
    #pragma unroll
    for (int i = 0; i < 8; ++i)
      hbuf[m * DD + c + i] = h0[(size_t)(r0 + m) * DD + c + i];
    if (tid < 16) y[(size_t)(r0 + tid) * SS + (SS - 1)] = 0.0f;
  }
  __syncthreads();

  for (int t = 0; t < SS - 1; ++t) {
#if HAVE_ASYNC_LDS
    {  // async prefetch x_{t+1} tile (16 rows x 128 bf16) into LDS
      int row = tid >> 4, off = (tid & 15) * 8;
      const __bf16* gsrc = xw + ((size_t)(r0 + row) * SS + (t + 1)) * DD + off;
      async_copy_b128(gsrc, xnb + row * DD + off);
    }
#endif
    __builtin_prefetch(wkis + (size_t)lane * 16, 0, 0);

    if (tid < 16) yred[tid] = 0.0f;
    {  // stage input_sdf = x_t - h  (K=128)
      int m = tid >> 4, c0 = (tid & 15) * 8;
      const __bf16* xr = xw + ((size_t)(r0 + m) * SS + t) * DD;
      #pragma unroll
      for (int i = 0; i < 8; ++i) {
        int k = c0 + i;
        float v = (float)xr[k] - hbuf[m * DD + k];
        act[a_index(0, k >> 5, m, k & 31)] = (__bf16)v;
      }
    }
    __syncthreads();

    // SDF: two GEMMs, K=128
    v8f a1 = {}, a2 = {};
    #pragma unroll
    for (int kt = 0; kt < 4; ++kt) {
      v16bf av = *(const v16bf*)(act + (kt * 32 + lane) * 16);
      v16bf b1 = *(const v16bf*)(wsdf1s + ((kt * 8 + nt) * 32 + lane) * 16);
      v16bf b2 = *(const v16bf*)(wsdf2s + ((kt * 8 + nt) * 32 + lane) * 16);
      a1 = wmma_bf16(av, b1, a1);
      a2 = wmma_bf16(av, b2, a2);
    }
    float sdf[8];
    {
      float q1 = bsdf1[n], q2 = bsdf2[n];
      #pragma unroll
      for (int r = 0; r < 8; ++r)
        sdf[r] = sigmoidf_(a1[r] + q1) * tanhf(a2[r] + q2);
    }
    __syncthreads();  // act (input_sdf) fully consumed

    {  // stage pka input: cols [0,128) = sdf (from C layout), [128,256) = co_t
      #pragma unroll
      for (int r = 0; r < 8; ++r)
        act[a_index(0, n >> 5, r + mhalf, n & 31)] = (__bf16)sdf[r];
      int m = tid >> 4, c0 = (tid & 15) * 8;
      const float* cr = Ecorr + (size_t)corr[(size_t)(r0 + m) * SS + t] * DD;
      #pragma unroll
      for (int i = 0; i < 8; ++i) {
        int k = DD + c0 + i;
        act[a_index(0, k >> 5, m, k & 31)] = (__bf16)cr[c0 + i];
      }
    }
    __syncthreads();

    // PKA: two GEMMs, K=256
    v8f p1 = {}, p2 = {};
    #pragma unroll
    for (int kt = 0; kt < 8; ++kt) {
      v16bf av = *(const v16bf*)(act + (kt * 32 + lane) * 16);
      v16bf b1 = *(const v16bf*)(wpka1s + ((kt * 8 + nt) * 32 + lane) * 16);
      v16bf b2 = *(const v16bf*)(wpka2s + ((kt * 8 + nt) * 32 + lane) * 16);
      p1 = wmma_bf16(av, b1, p1);
      p2 = wmma_bf16(av, b2, p2);
    }
    float pka[8];
    {
      float q1 = bpka1[n], q2 = bpka2[n];
      #pragma unroll
      for (int r = 0; r < 8; ++r)
        pka[r] = sigmoidf_(p1[r] + q1) * tanhf(p2[r] + q2);
    }
    __syncthreads();  // act (pka input) fully consumed

    {  // stage ki input: [h | co | qd | cd]  (K=512)
      int m = tid >> 4, c0 = (tid & 15) * 32;
      int seg = c0 >> 7;  // 32-wide blocks never cross a 128 boundary
      const float* src = nullptr;
      if (seg == 1)
        src = Ecorr + (size_t)corr[(size_t)(r0 + m) * SS + t] * DD + (c0 - DD);
      else if (seg == 2)
        src = Eqd + (size_t)qds[(size_t)(r0 + m) * SS + t] * DD + (c0 - 2 * DD);
      else if (seg == 3)
        src = Ecd + (size_t)cds[(size_t)(r0 + m) * SS + t] * DD + (c0 - 3 * DD);
      #pragma unroll
      for (int i = 0; i < 32; ++i) {
        int k = c0 + i;
        float v = (seg == 0) ? hbuf[m * DD + k] : src[i];
        act[a_index(0, k >> 5, m, k & 31)] = (__bf16)v;
      }
    }
#if HAVE_ASYNC_LDS
    wait_async0();
#endif
    __syncthreads();

    // KI: one GEMM, K=512
    v8f gac = {};
    #pragma unroll
    for (int kt = 0; kt < 16; ++kt) {
      v16bf av = *(const v16bf*)(act + (kt * 32 + lane) * 16);
      v16bf bw = *(const v16bf*)(wkis + ((kt * 8 + nt) * 32 + lane) * 16);
      gac = wmma_bf16(av, bw, gac);
    }

    // h update + partial dot with x_{t+1} (each (m,n) owned by one lane)
    {
      float qb = bki[n];
      #pragma unroll
      for (int r = 0; r < 8; ++r) {
        int m = r + mhalf;
        float gamma = sigmoidf_(gac[r] + qb);
        float hnew = gamma * hbuf[m * DD + n] + (1.0f - gamma) * pka[r];
        hbuf[m * DD + n] = hnew;
#if HAVE_ASYNC_LDS
        float xv = (float)xnb[m * DD + n];
#else
        float xv = (float)xw[((size_t)(r0 + m) * SS + (t + 1)) * DD + n];
#endif
        atomicAdd(&yred[m], hnew * xv);
      }
    }
    __syncthreads();
    if (tid < 16) y[(size_t)(r0 + tid) * SS + t] = sigmoidf_(yred[tid]);
    __syncthreads();
  }
}

// ---------------------------------------------------------------------------
extern "C" void kernel_launch(void* const* d_in, const int* in_sizes, int n_in,
                              void* d_out, int out_size, void* d_ws, size_t ws_size,
                              hipStream_t stream) {
  const int* qs    = (const int*)d_in[0];
  const int* cs    = (const int*)d_in[1];
  const int* qdseq = (const int*)d_in[2];
  const int* cdseq = (const int*)d_in[3];
  const int* corr  = (const int*)d_in[4];
  const float* Eq    = (const float*)d_in[5];
  const float* Ec    = (const float*)d_in[6];
  const float* Eqd   = (const float*)d_in[7];
  const float* Ecd   = (const float*)d_in[8];
  const float* Ecorr = (const float*)d_in[9];
  const float* Wx    = (const float*)d_in[10]; const float* bx    = (const float*)d_in[11];
  const float* Wsdf1 = (const float*)d_in[12]; const float* bsdf1 = (const float*)d_in[13];
  const float* Wsdf2 = (const float*)d_in[14]; const float* bsdf2 = (const float*)d_in[15];
  const float* Wpka1 = (const float*)d_in[16]; const float* bpka1 = (const float*)d_in[17];
  const float* Wpka2 = (const float*)d_in[18]; const float* bpka2 = (const float*)d_in[19];
  const float* Wki   = (const float*)d_in[20]; const float* bki   = (const float*)d_in[21];
  const float* h0    = (const float*)d_in[22];
  float* y = (float*)d_out;

  // workspace layout (bf16): x_all [B*S, D], then swizzled weights
  __bf16* xw  = (__bf16*)d_ws;
  __bf16* wws = xw + (size_t)BB * SS * DD;
  __bf16* wxs    = wws;            // 512*128
  __bf16* wsdf1s = wws + 65536;    // 128*128
  __bf16* wsdf2s = wws + 81920;    // 128*128
  __bf16* wpka1s = wws + 98304;    // 256*128
  __bf16* wpka2s = wws + 131072;   // 256*128
  __bf16* wkis   = wws + 163840;   // 512*128

  swz_w<<<(512 * 128 + 255) / 256, 256, 0, stream>>>(Wx,    wxs,    512);
  swz_w<<<(128 * 128 + 255) / 256, 256, 0, stream>>>(Wsdf1, wsdf1s, 128);
  swz_w<<<(128 * 128 + 255) / 256, 256, 0, stream>>>(Wsdf2, wsdf2s, 128);
  swz_w<<<(256 * 128 + 255) / 256, 256, 0, stream>>>(Wpka1, wpka1s, 256);
  swz_w<<<(256 * 128 + 255) / 256, 256, 0, stream>>>(Wpka2, wpka2s, 256);
  swz_w<<<(512 * 128 + 255) / 256, 256, 0, stream>>>(Wki,   wkis,   512);

  xgemm_kernel<<<(BB * SS) / 64, 256, 0, stream>>>(qs, cs, qdseq, cdseq,
                                                   Eq, Ec, Eqd, Ecd,
                                                   wxs, bx, xw);

  scan_kernel<<<BB / 16, 256, 0, stream>>>(xw, wsdf1s, wsdf2s, wpka1s, wpka2s,
                                           wkis, bsdf1, bsdf2, bpka1, bpka2, bki,
                                           corr, qdseq, cdseq, Ecorr, Eqd, Ecd,
                                           h0, y);
}
